// Multi_Head_Attention_18176301596914
// MI455X (gfx1250) — compile-verified, matrix operand order checked by reading, not run on hardware
//
#include <hip/hip_runtime.h>
#include <hip/hip_bf16.h>

// MI455X / gfx1250, wave32. All matrix math via v_wmma_f32_16x16x32_bf16.
// Requires ws_size >= 32 MiB (4 x [4096 x 1024] bf16 buffers).

typedef __attribute__((ext_vector_type(16))) __bf16 v16bf;
typedef __attribute__((ext_vector_type(8)))  float  v8f;
static_assert(sizeof(v16bf) == 32, "v16bf must be 32 bytes");

#define DMODEL 1024
#define NHEADS 16
#define HDIM   64
#define BB     2
#define SS     2048
#define MROWS  (BB * SS)   // 4096

// CDNA5 async global->LDS copy path (ASYNCcnt), with safe fallback.
// Builtin signature (from hipcc diagnostics): pointee type is a 16-byte int
// vector; global side is AS(1), LDS side is AS(3).
#if defined(__AMDGCN__) && defined(__gfx1250__) && \
    __has_builtin(__builtin_amdgcn_global_load_async_to_lds_b128) && \
    __has_builtin(__builtin_amdgcn_s_wait_asynccnt)
#define USE_ASYNC_LDS 1
typedef int v4i_vs __attribute__((vector_size(16)));
#define GLB_PTR(p)  ((__attribute__((address_space(1))) v4i_vs*)(const_cast<unsigned short*>(p)))
#define LDS_PTR(p)  ((__attribute__((address_space(3))) v4i_vs*)(p))
#else
#define USE_ASYNC_LDS 0
#endif

__device__ __forceinline__ unsigned short f2bf(float f) {
  union { __bf16 h; unsigned short u; } v;
  v.h = (__bf16)f;                 // native f32->bf16 cvt (v_cvt_pk_bf16_f32)
  return v.u;
}

// Build a 16x32 bf16 WMMA fragment for one lane from two contiguous 16-byte
// chunks. Layout (ISA 7.12.2, 16-bit A 16x32): lane holds row m=lane&15;
// half=lane>>4; elements 0..7 <-> k = 8*half..8*half+7 ;
// elements 8..15 <-> k = 16+8*half..16+8*half+7. B mirrors with n=lane&15.
__device__ __forceinline__ v16bf make_frag(const unsigned short* p_lo,
                                           const unsigned short* p_hi) {
  union { uint4 u[2]; v16bf v; } f;
  f.u[0] = *(const uint4*)p_lo;
  f.u[1] = *(const uint4*)p_hi;
  return f.v;
}

// ---------------------------------------------------------------------------
// GEMM: C[M=4096, N=1024] = A[M,K=1024] * W[K,N] + bias   (then * out_scale)
//   A_BF16=false : A fp32, converted to bf16 on stage
//   A_BF16=true  : A already bf16 -> staged with async B128 LDS copies
//   HEADSPLIT=true : emit bf16 into [B,H,S,Dh] layout (QKV projections)
//   HEADSPLIT=false: emit fp32 row-major (final projection -> d_out)
// Block 256 threads = 8 waves; block tile 256x128; wave tile 64x64
// (16 WMMA per 16 ds_load_b128 per K-step).
// ---------------------------------------------------------------------------
template<bool A_BF16, bool HEADSPLIT>
__global__ __launch_bounds__(256) void gemm_bias_wmma(
    const float* __restrict__ A, const unsigned short* __restrict__ Abf,
    const float* __restrict__ W, const float* __restrict__ bias,
    float* __restrict__ outF, unsigned short* __restrict__ outB,
    float out_scale)
{
  constexpr int BM = 256, BN = 128, BK = 32, K = DMODEL, N = DMODEL;
  __shared__ __align__(16) unsigned short sA[BM][BK];   // 16 KB, [m][k]
  __shared__ __align__(16) unsigned short sB[BN][BK];   //  8 KB, [n][k] (W^T)

  const int tid  = threadIdx.x;
  const int lane = tid & 31;
  const int wave = tid >> 5;           // 8 waves
  const int m0 = blockIdx.y * BM;
  const int n0 = blockIdx.x * BN;
  const int wm = (wave >> 1) * 64;     // 4 wave-rows of 64
  const int wn = (wave & 1) * 64;      // 2 wave-cols of 64
  const int lm = lane & 15;
  const int half = lane >> 4;

  v8f acc[4][4] = {};

  for (int k0 = 0; k0 < K; k0 += BK) {
    // ---- stage A tile (256x32) as bf16 ----
    if (A_BF16) {
      // straight bf16 copy: 1024 16-byte chunks, 4 per thread
      #pragma unroll
      for (int i = 0; i < 4; ++i) {
        int idx = tid + 256 * i;
        int row = idx >> 2;
        int c   = (idx & 3) * 8;
        const unsigned short* src = Abf + (size_t)(m0 + row) * K + k0 + c;
#if USE_ASYNC_LDS
        __builtin_amdgcn_global_load_async_to_lds_b128(
            GLB_PTR(src), LDS_PTR(&sA[row][c]), 0, 0);
#else
        *(uint4*)&sA[row][c] = *(const uint4*)src;
#endif
      }
    } else {
      // fp32 -> bf16 convert while staging, packed pairs along k
      #pragma unroll
      for (int i = 0; i < 16; ++i) {
        int idx = tid + 256 * i;       // 4096 element-pairs
        int row = idx >> 4;
        int kp  = idx & 15;
        float2 f = *(const float2*)(A + (size_t)(m0 + row) * K + k0 + 2 * kp);
        sA[row][2 * kp]     = f2bf(f.x);
        sA[row][2 * kp + 1] = f2bf(f.y);
      }
    }
    // ---- stage W tile transposed: sB[n][k] = bf16(W[k0+k][n0+n]) ----
    #pragma unroll
    for (int i = 0; i < 16; ++i) {
      int idx = tid + 256 * i;         // 4096 elements, coalesced over n
      int n  = idx & 127;
      int kk = idx >> 7;
      sB[n][kk] = f2bf(W[(size_t)(k0 + kk) * N + n0 + n]);
    }
#if USE_ASYNC_LDS
    if (A_BF16) __builtin_amdgcn_s_wait_asynccnt(0);  // LDS writes visible
#endif
    __syncthreads();

    v16bf af[4], bfr[4];
    #pragma unroll
    for (int i = 0; i < 4; ++i) {
      const unsigned short* r = &sA[wm + 16 * i + lm][0];
      af[i] = make_frag(r + 8 * half, r + 16 + 8 * half);
    }
    #pragma unroll
    for (int j = 0; j < 4; ++j) {
      const unsigned short* r = &sB[wn + 16 * j + lm][0];
      bfr[j] = make_frag(r + 8 * half, r + 16 + 8 * half);
    }
    #pragma unroll
    for (int i = 0; i < 4; ++i)
      #pragma unroll
      for (int j = 0; j < 4; ++j)
        acc[i][j] = __builtin_amdgcn_wmma_f32_16x16x32_bf16(
            false, af[i], false, bfr[j], (short)0, acc[i][j], false, false);
    __syncthreads();
  }

  // ---- epilogue: C layout = lane n=lane&15, rows r + 8*half ----
  #pragma unroll
  for (int i = 0; i < 4; ++i) {
    #pragma unroll
    for (int j = 0; j < 4; ++j) {
      #pragma unroll
      for (int r = 0; r < 8; ++r) {
        int m = m0 + wm + 16 * i + 8 * half + r;
        int n = n0 + wn + 16 * j + lm;
        float v = acc[i][j][r] + bias[n];
        if (HEADSPLIT) {
          int b = m >> 11, s = m & (SS - 1);
          int h = n >> 6,  d = n & (HDIM - 1);
          outB[(((size_t)b * NHEADS + h) * SS + s) * HDIM + d] =
              f2bf(v * out_scale);
        } else {
          outF[(size_t)m * N + n] = v;
        }
      }
    }
  }
}

// ---------------------------------------------------------------------------
// Flash attention, TRANSPOSED formulation:
//   S^T = K . Q^T   (A = K-tile rows, B = Q fragment; the Q A-fragment is
//                    bit-identical to the Q^T B-fragment)
//   O^T += V^T . P^T (A = V^T from LDS, B = P^T packed in registers: the S^T
//                    C-register layout IS the P^T B-fragment layout, so the
//                    softmax result feeds WMMA with no LDS round-trip)
// Each lane owns ONE q-column => softmax stats are per-lane scalars and need
// only a single xor-16 shuffle per reduction. 1/sqrt(Dh) is pre-folded into
// the Q projection. K tile staged via async global->LDS (ASYNCcnt) when
// available; V staged through VGPRs because it is transposed on the fly.
// q/k/v: bf16 [B,H,S,64]; output: bf16 [B,S,1024] (col = h*64 + d).
// ---------------------------------------------------------------------------
__global__ __launch_bounds__(256) void flash_attn_wmma(
    const unsigned short* __restrict__ qb,
    const unsigned short* __restrict__ kb,
    const unsigned short* __restrict__ vb,
    unsigned short* __restrict__ ab)
{
  __shared__ __align__(16) unsigned short sK[32][64];   // 4 KB  [krow][d]
  __shared__ __align__(16) unsigned short sV[64][32];   // 4 KB  [d][krow]

  const int tid  = threadIdx.x;
  const int lane = tid & 31;
  const int wave = tid >> 5;
  const int lm   = lane & 15;
  const int half = lane >> 4;
  const int bh = blockIdx.y;
  const int b = bh >> 4, h = bh & 15;
  const int qrow0 = blockIdx.x * 128 + wave * 16;
  const size_t head_base = (size_t)bh * SS * HDIM;

  // Q^T B-fragments: 64 features x 16 q-cols -> two 16x32-shaped fragments.
  // Per-lane data identical to the Q A-fragment: lane=q-row, elements=d.
  v16bf bq[2];
  {
    const unsigned short* qp = qb + head_base + (size_t)(qrow0 + lm) * HDIM;
    #pragma unroll
    for (int kk = 0; kk < 2; ++kk)
      bq[kk] = make_frag(qp + 32 * kk + 8 * half,
                         qp + 32 * kk + 16 + 8 * half);
  }

  float mst = -3.0e38f;   // running max for q-col = lane&15
  float lst = 0.0f;       // running sum
  v8f accO[4] = {};       // O^T: tile j = d-rows 16j..16j+15 x 16 q-cols

  const int srow = tid >> 3;          // staging: row 0..31
  const int scol = (tid & 7) * 8;     // staging: d 0..56 step 8

  for (int kt = 0; kt < SS / 32; ++kt) {
    const int kr0 = kt * 32;
    const size_t tile_base = head_base + (size_t)(kr0 + srow) * HDIM + scol;
    // ---- stage K row-major (async copy when available) ----
#if USE_ASYNC_LDS
    __builtin_amdgcn_global_load_async_to_lds_b128(
        GLB_PTR(kb + tile_base), LDS_PTR(&sK[srow][scol]), 0, 0);
#else
    *(uint4*)&sK[srow][scol] = *(const uint4*)(kb + tile_base);
#endif
    // ---- stage V transposed through VGPRs ----
    {
      union { uint4 q; unsigned short s[8]; } u;
      u.q = *(const uint4*)(vb + tile_base);
      #pragma unroll
      for (int e = 0; e < 8; ++e) sV[scol + e][srow] = u.s[e];
    }
#if USE_ASYNC_LDS
    __builtin_amdgcn_s_wait_asynccnt(0);   // our async LDS writes landed
#endif
    __syncthreads();

    // ---- S^T = K . Q^T : two 16x16 C tiles (k-rows 0..15 / 16..31) ----
    v8f sc[2] = {};
    #pragma unroll
    for (int n = 0; n < 2; ++n) {
      const unsigned short* kr = &sK[16 * n + lm][0];
      #pragma unroll
      for (int kk = 0; kk < 2; ++kk) {
        v16bf aK = make_frag(kr + 32 * kk + 8 * half,
                             kr + 32 * kk + 16 + 8 * half);
        sc[n] = __builtin_amdgcn_wmma_f32_16x16x32_bf16(
            false, aK, false, bq[kk], (short)0, sc[n], false, false);
      }
    }

    // ---- online softmax over the 32 k-scores of this lane's q-column ----
    // Lane holds sc[n][r] <-> k = 16n + r + 8*half; partner lane (xor 16)
    // holds the other 16. One shuffle per reduction.
    float t = fmaxf(sc[0][0], sc[1][0]);
    #pragma unroll
    for (int r = 1; r < 8; ++r) t = fmaxf(t, fmaxf(sc[0][r], sc[1][r]));
    t = fmaxf(t, __shfl_xor(t, 16, 32));
    const float mn   = fmaxf(mst, t);
    const float corr = __expf(mst - mn);

    // exp() and pack P^T directly into a B-fragment: element r <-> k=8h+r
    // (tile 0), element 8+r <-> k=16+8h+r (tile 1) == S^T C layout.
    union { unsigned short us[16]; v16bf v; } pf;
    float rs = 0.0f;
    #pragma unroll
    for (int r = 0; r < 8; ++r) {
      float p0 = __expf(sc[0][r] - mn);
      float p1 = __expf(sc[1][r] - mn);
      rs += p0 + p1;
      pf.us[r]     = f2bf(p0);
      pf.us[8 + r] = f2bf(p1);
    }
    rs += __shfl_xor(rs, 16, 32);
    lst = lst * corr + rs;
    mst = mn;

    #pragma unroll
    for (int j = 0; j < 4; ++j)
      #pragma unroll
      for (int r = 0; r < 8; ++r) accO[j][r] *= corr;

    // ---- O^T += V^T . P^T ----
    #pragma unroll
    for (int j = 0; j < 4; ++j) {
      const unsigned short* vr = &sV[16 * j + lm][0];
      v16bf aV = make_frag(vr + 8 * half, vr + 16 + 8 * half);
      accO[j] = __builtin_amdgcn_wmma_f32_16x16x32_bf16(
          false, aV, false, pf.v, (short)0, accO[j], false, false);
    }
    __syncthreads();
  }

  // ---- normalize; emit 8 consecutive d per (j,half) as one 16B store ----
  const float inv = 1.0f / lst;
  const int s = qrow0 + lm;            // this lane's q row
  unsigned short* orow = ab + ((size_t)b * SS + s) * DMODEL + h * HDIM;
  #pragma unroll
  for (int j = 0; j < 4; ++j) {
    union { unsigned short us[8]; uint4 u; } o;
    #pragma unroll
    for (int r = 0; r < 8; ++r) o.us[r] = f2bf(accO[j][r] * inv);
    *(uint4*)(orow + 16 * j + 8 * half) = o.u;   // 16B aligned
  }
}

extern "C" void kernel_launch(void* const* d_in, const int* in_sizes, int n_in,
                              void* d_out, int out_size, void* d_ws, size_t ws_size,
                              hipStream_t stream) {
  (void)in_sizes; (void)n_in; (void)out_size; (void)ws_size;
  const float* Q  = (const float*)d_in[0];
  const float* Kx = (const float*)d_in[1];
  const float* V  = (const float*)d_in[2];
  const float* Wq = (const float*)d_in[3];
  const float* bq = (const float*)d_in[4];
  const float* Wk = (const float*)d_in[5];
  const float* bk = (const float*)d_in[6];
  const float* Wv = (const float*)d_in[7];
  const float* bv = (const float*)d_in[8];
  const float* Wo = (const float*)d_in[9];
  const float* bo = (const float*)d_in[10];
  float* out = (float*)d_out;

  unsigned short* qbuf = (unsigned short*)d_ws;                 // [B,H,S,64] bf16
  unsigned short* kbuf = qbuf + (size_t)MROWS * DMODEL;
  unsigned short* vbuf = kbuf + (size_t)MROWS * DMODEL;
  unsigned short* abuf = vbuf + (size_t)MROWS * DMODEL;         // [B,S,D] bf16

  const float qscale = 0.125f;   // 1/sqrt(HDIM), folded into Q projection

  dim3 gG(DMODEL / 128, MROWS / 256);   // (8, 16)
  gemm_bias_wmma<false, true><<<gG, 256, 0, stream>>>(Q,  nullptr, Wq, bq, nullptr, qbuf, qscale);
  gemm_bias_wmma<false, true><<<gG, 256, 0, stream>>>(Kx, nullptr, Wk, bk, nullptr, kbuf, 1.0f);
  gemm_bias_wmma<false, true><<<gG, 256, 0, stream>>>(V,  nullptr, Wv, bv, nullptr, vbuf, 1.0f);

  dim3 gA(SS / 128, BB * NHEADS);       // (16, 32)
  flash_attn_wmma<<<gA, 256, 0, stream>>>(qbuf, kbuf, vbuf, abuf);

  gemm_bias_wmma<true, false><<<gG, 256, 0, stream>>>(nullptr, abuf, Wo, bo, out, nullptr, 1.0f);
}
